// SelfAttnLayer_56418690400264
// MI455X (gfx1250) — matrix operand order repaired and checked by reading, not compiled, not run on hardware
//
#include <hip/hip_runtime.h>

// ---------------------------------------------------------------------------
// Self-attention layer for MI455X (gfx1250), bf16 WMMA + TDM async tile loads.
// B=2, S=2048, DIM=1024, HEADS=16, HEAD_DIM=64, M = B*S = 4096.
// ---------------------------------------------------------------------------

typedef __attribute__((ext_vector_type(16))) __bf16 v16bf;
typedef __attribute__((ext_vector_type(8)))  float  v8f;
typedef unsigned int u32x4 __attribute__((ext_vector_type(4)));
typedef int          i32x4 __attribute__((ext_vector_type(4)));
typedef int          i32x8 __attribute__((ext_vector_type(8)));

union Frag {
    v16bf v;
    uint4 q[2];
    unsigned short u[16];
};

__device__ __forceinline__ unsigned short f2bf(float x) {
    unsigned int b = __float_as_uint(x);
    b += 0x7FFFu + ((b >> 16) & 1u);   // round-to-nearest-even
    return (unsigned short)(b >> 16);
}

// ---------------------------------------------------------------------------
// TDM: DMA a [tile1 x tile0] bf16 tile (row stride `stride0` elements) from
// global memory into LDS at `lds_ptr`, padding rows 32 data halfs -> 40 halfs
// (pad_interval code 3 = 16 DWORDs, pad_amount code 3 = 4 DWORDs).
// Issued by one wave; completion via TENSORcnt.
// ---------------------------------------------------------------------------
#if __has_builtin(__builtin_amdgcn_tensor_load_to_lds)
#define HAVE_TDM 1
#else
#define HAVE_TDM 0
#endif

__device__ __forceinline__ void tdm_issue_tile(
    void* lds_ptr, const unsigned short* src,
    unsigned int dim0, unsigned int dim1, unsigned int stride0,
    unsigned int tile0, unsigned int tile1)
{
#if HAVE_TDM
    unsigned long long ga = (unsigned long long)(uintptr_t)src;
    unsigned int lds_addr = (unsigned int)(uintptr_t)lds_ptr;   // low 32 = LDS offset
    u32x4 g0;
    g0[0] = 1u;                                         // count=1 (valid), user mode
    g0[1] = lds_addr;                                   // lds_addr
    g0[2] = (unsigned int)(ga & 0xFFFFFFFFu);           // global_addr[31:0]
    g0[3] = (unsigned int)((ga >> 32) & 0x1FFFFFFu) | (2u << 30);  // [56:32] + type=2
    i32x8 g1;
    g1[0] = (int)((1u << 16)            // data_size = 2 bytes
                | (1u << 20)            // pad_enable
                | (3u << 22)            // pad_interval: 16 DWORDs
                | (3u << 25));          // pad_amount:   4 DWORDs
    g1[1] = (int)((dim0 & 0xFFFFu) << 16);                            // tensor_dim0 lo
    g1[2] = (int)(((dim0 >> 16) & 0xFFFFu) | ((dim1 & 0xFFFFu) << 16));
    g1[3] = (int)(((dim1 >> 16) & 0xFFFFu) | ((tile0 & 0xFFFFu) << 16));
    g1[4] = (int)(tile1 & 0xFFFFu);                                   // tile_dim1, tile_dim2=0
    g1[5] = (int)stride0;                                             // tensor_dim0_stride lo
    g1[6] = 0;
    g1[7] = 0;
    i32x4 z4 = {0, 0, 0, 0};
#if __clang_major__ >= 23
    i32x8 z8 = {0, 0, 0, 0, 0, 0, 0, 0};
    __builtin_amdgcn_tensor_load_to_lds(g0, g1, z4, z4, z8, 0);
#else
    __builtin_amdgcn_tensor_load_to_lds(g0, g1, z4, z4, 0);
#endif
#else
    // Fallback: issuing wave (32 lanes) copies the tile manually.
    unsigned short* dst = (unsigned short*)lds_ptr;
    int lane = threadIdx.x & 31;
    for (int r = lane; r < (int)tile1; r += 32) {
        const uint4* s = (const uint4*)(src + (size_t)r * stride0);
        uint4* d = (uint4*)(dst + (size_t)r * 40);
        for (unsigned int cg = 0; cg < tile0 / 8; ++cg) d[cg] = s[cg];
    }
#endif
}

__device__ __forceinline__ void tdm_wait(int pending) {
#if HAVE_TDM
#if __has_builtin(__builtin_amdgcn_s_wait_tensorcnt)
    if (pending) __builtin_amdgcn_s_wait_tensorcnt((short)1);
    else         __builtin_amdgcn_s_wait_tensorcnt((short)0);
#endif
#else
    (void)pending;
#endif
}

// ---------------------------------------------------------------------------
// fp32 -> bf16 conversion
// ---------------------------------------------------------------------------
__global__ void attn_f32_to_bf16(const float* __restrict__ x,
                                 unsigned short* __restrict__ y, int n) {
    int i = blockIdx.x * blockDim.x + threadIdx.x;
    if (i < n) y[i] = f2bf(x[i]);
}

// ---------------------------------------------------------------------------
// bf16 GEMM: C[M,N] = scale * (A[M,K] @ B[K,N]), f32 accumulate.
// 256 threads (8 waves), workgroup tile 64(M) x 128(N); wave does 16x64.
// A tile double-buffered via TDM; B tile staged transposed (Bs[n][k]).
// mode: 0 = f32 row-major, 1 = bf16 row-major, 2 = bf16 V-transposed
//       (out[(gm>>11)*1024 + gn][gm & 2047] in a [B*HEADS*64, S] buffer).
// ---------------------------------------------------------------------------
__global__ __launch_bounds__(256) void attn_gemm_bf16(
    const unsigned short* __restrict__ A,
    const unsigned short* __restrict__ Bm,
    void* __restrict__ Cout,
    int M, int N, int K, float scale, int mode)
{
    __shared__ __align__(16) unsigned short As[2][64][40];  // [buf][m][k]
    __shared__ __align__(16) unsigned short Bs[128][40];    // transposed: [n][k]

    const int tid  = threadIdx.x;
    const int wave = tid >> 5;
    const int lane = tid & 31;
    const int wm   = wave >> 1;        // 0..3  (16-row M sub-tile)
    const int wn   = wave & 1;         // 0..1  (64-col N half)
    const int h    = lane >> 4;
    const int n16  = lane & 15;

    const int bm = blockIdx.x * 64;
    const int bn = blockIdx.y * 128;

    v8f acc[4] = {{}, {}, {}, {}};

    const int nIter = K / 32;
    // Prologue: DMA first A tile into buffer 0.
    if (tid < 32)
        tdm_issue_tile(&As[0][0][0], A + (size_t)bm * K,
                       (unsigned)K, (unsigned)M, (unsigned)K, 32u, 64u);

    for (int it = 0; it < nIter; ++it) {
        const int k0  = it * 32;
        const int cur = it & 1;

        // Stage B tile 32x128, transposed into Bs[n][k].
        {
            int kr = tid >> 3;            // 0..31
            int nc = (tid & 7) * 16;      // 0,16,...,112
            const unsigned short* src = Bm + (size_t)(k0 + kr) * N + bn + nc;
            uint4 d0 = *(const uint4*)(src);
            uint4 d1 = *(const uint4*)(src + 8);
            if (k0 + 32 < K)
                __builtin_prefetch(Bm + (size_t)(k0 + 32 + kr) * N + bn + nc, 0, 1);
            const unsigned short* p0 = (const unsigned short*)&d0;
            const unsigned short* p1 = (const unsigned short*)&d1;
            #pragma unroll
            for (int j = 0; j < 8; ++j) Bs[nc + j][kr]     = p0[j];
            #pragma unroll
            for (int j = 0; j < 8; ++j) Bs[nc + 8 + j][kr] = p1[j];
        }

        // Issue next A tile, then wait for the current one.
        if (tid < 32) {
            if (it + 1 < nIter) {
                tdm_issue_tile(&As[cur ^ 1][0][0], A + (size_t)bm * K + k0 + 32,
                               (unsigned)K, (unsigned)M, (unsigned)K, 32u, 64u);
                tdm_wait(1);
            } else {
                tdm_wait(0);
            }
        }
        __syncthreads();

        Frag fa;
        fa.q[0] = *(const uint4*)&As[cur][wm * 16 + n16][16 * h];
        fa.q[1] = *(const uint4*)&As[cur][wm * 16 + n16][16 * h + 8];
        #pragma unroll
        for (int c = 0; c < 4; ++c) {
            Frag fb;
            fb.q[0] = *(const uint4*)&Bs[wn * 64 + c * 16 + n16][16 * h];
            fb.q[1] = *(const uint4*)&Bs[wn * 64 + c * 16 + n16][16 * h + 8];
            acc[c] = __builtin_amdgcn_wmma_f32_16x16x32_bf16(
                         false, fa.v, false, fb.v, (short)0, acc[c], false, false);
        }
        __syncthreads();
    }

    // Epilogue: c[j] = C[8h+j][n]
    #pragma unroll
    for (int c = 0; c < 4; ++c) {
        #pragma unroll
        for (int j = 0; j < 8; ++j) {
            int gm = bm + wm * 16 + 8 * h + j;
            int gn = bn + wn * 64 + c * 16 + n16;
            float v = acc[c][j] * scale;
            if (mode == 0) {
                ((float*)Cout)[(size_t)gm * N + gn] = v;
            } else if (mode == 1) {
                ((unsigned short*)Cout)[(size_t)gm * N + gn] = f2bf(v);
            } else {
                // V-transpose: [b*HEADS*64 + gn][s] with s = gm & 2047
                size_t idx = ((size_t)(gm >> 11) * 1024 + gn) * 2048 + (gm & 2047);
                ((unsigned short*)Cout)[idx] = f2bf(v);
            }
        }
    }
}

// ---------------------------------------------------------------------------
// Flash attention: one wave per 16-query tile per (b, head).
// Q pre-scaled by HEAD_DIM^-0.5. Online softmax (f32 stats), bf16 WMMA.
// Q/K: bf16 [4096,1024] row-major. Vt: bf16 [B*HEADS*64, 2048] (d-major).
// ---------------------------------------------------------------------------
__global__ __launch_bounds__(32) void attn_flash(
    const unsigned short* __restrict__ Qb,
    const unsigned short* __restrict__ Kb,
    const unsigned short* __restrict__ Vt,
    const int* __restrict__ mask,
    unsigned short* __restrict__ Ob)
{
    __shared__ __align__(16) unsigned short Pl[16][40];  // P tile: [q][key]

    const int bh = blockIdx.x;            // 0..31
    const int b  = bh >> 4;
    const int hh = bh & 15;
    const int q0 = blockIdx.y * 16;
    const int lane = threadIdx.x;
    const int h  = lane >> 4;
    const int n  = lane & 15;

    // Resident Q fragments (d 0..31 and 32..63)
    Frag aq0, aq1;
    {
        size_t rq = (size_t)(b * 2048 + q0 + n) * 1024 + hh * 64;
        aq0.q[0] = *(const uint4*)(Qb + rq + 16 * h);
        aq0.q[1] = *(const uint4*)(Qb + rq + 16 * h + 8);
        aq1.q[0] = *(const uint4*)(Qb + rq + 32 + 16 * h);
        aq1.q[1] = *(const uint4*)(Qb + rq + 32 + 16 * h + 8);
    }

    float mrow[8], lrow[8];
    v8f o[4] = {{}, {}, {}, {}};
    #pragma unroll
    for (int j = 0; j < 8; ++j) { mrow[j] = -3.0e38f; lrow[j] = 0.0f; }

    const float FMIN = -3.4028235e38f;

    for (int kb = 0; kb < 2048; kb += 32) {
        float add0 = (1.0f - (float)mask[b * 2048 + kb + n])      * FMIN;
        float add1 = (1.0f - (float)mask[b * 2048 + kb + 16 + n]) * FMIN;

        // K fragments: b[i] = K[key][d=32t+16h+i]; key = kb + 16c + n
        Frag bk00, bk10, bk01, bk11;
        {
            size_t r0 = (size_t)(b * 2048 + kb + n)      * 1024 + hh * 64;
            size_t r1 = (size_t)(b * 2048 + kb + 16 + n) * 1024 + hh * 64;
            bk00.q[0] = *(const uint4*)(Kb + r0 + 16 * h);
            bk00.q[1] = *(const uint4*)(Kb + r0 + 16 * h + 8);
            bk10.q[0] = *(const uint4*)(Kb + r0 + 32 + 16 * h);
            bk10.q[1] = *(const uint4*)(Kb + r0 + 32 + 16 * h + 8);
            bk01.q[0] = *(const uint4*)(Kb + r1 + 16 * h);
            bk01.q[1] = *(const uint4*)(Kb + r1 + 16 * h + 8);
            bk11.q[0] = *(const uint4*)(Kb + r1 + 32 + 16 * h);
            bk11.q[1] = *(const uint4*)(Kb + r1 + 32 + 16 * h + 8);
        }

        v8f s0 = {}, s1 = {};
        s0 = __builtin_amdgcn_wmma_f32_16x16x32_bf16(
                 false, aq0.v, false, bk00.v, (short)0, s0, false, false);
        s0 = __builtin_amdgcn_wmma_f32_16x16x32_bf16(
                 false, aq1.v, false, bk10.v, (short)0, s0, false, false);
        s1 = __builtin_amdgcn_wmma_f32_16x16x32_bf16(
                 false, aq0.v, false, bk01.v, (short)0, s1, false, false);
        s1 = __builtin_amdgcn_wmma_f32_16x16x32_bf16(
                 false, aq1.v, false, bk11.v, (short)0, s1, false, false);

        // Online softmax update (row m = 8h+j lives across the 16 lanes of h)
        float p0[8], p1[8], sc[8];
        #pragma unroll
        for (int j = 0; j < 8; ++j) {
            float a0 = s0[j] + add0;
            float a1 = s1[j] + add1;
            float t = fmaxf(a0, a1);
            #pragma unroll
            for (int ofs = 1; ofs < 16; ofs <<= 1)
                t = fmaxf(t, __shfl_xor(t, ofs, 32));
            float mn  = fmaxf(mrow[j], t);
            float scj = __expf(mrow[j] - mn);
            p0[j] = __expf(a0 - mn);
            p1[j] = __expf(a1 - mn);
            float rs = p0[j] + p1[j];
            #pragma unroll
            for (int ofs = 1; ofs < 16; ofs <<= 1)
                rs += __shfl_xor(rs, ofs, 32);
            lrow[j] = lrow[j] * scj + rs;
            mrow[j] = mn;
            sc[j]   = scj;
        }
        #pragma unroll
        for (int t = 0; t < 4; ++t)
            #pragma unroll
            for (int j = 0; j < 8; ++j) o[t][j] *= sc[j];

        // P (C layout) -> LDS -> A layout
        #pragma unroll
        for (int j = 0; j < 8; ++j) {
            Pl[8 * h + j][n]      = f2bf(p0[j]);
            Pl[8 * h + j][16 + n] = f2bf(p1[j]);
        }
        __syncthreads();

        Frag pa;
        pa.q[0] = *(const uint4*)&Pl[n][16 * h];
        pa.q[1] = *(const uint4*)&Pl[n][16 * h + 8];

        // V fragments from transposed V: b[i] = Vt[d=16t+n][key=kb+16h+i]
        #pragma unroll
        for (int t = 0; t < 4; ++t) {
            Frag bv;
            size_t rv = ((size_t)((b * 16 + hh) * 64 + 16 * t + n)) * 2048
                        + kb + 16 * h;
            bv.q[0] = *(const uint4*)(Vt + rv);
            bv.q[1] = *(const uint4*)(Vt + rv + 8);
            o[t] = __builtin_amdgcn_wmma_f32_16x16x32_bf16(
                       false, pa.v, false, bv.v, (short)0, o[t], false, false);
        }
        __syncthreads();
    }

    // Epilogue: normalize and store O bf16 (row-major [4096,1024])
    #pragma unroll
    for (int j = 0; j < 8; ++j) {
        float inv = (lrow[j] > 0.0f) ? (1.0f / lrow[j]) : 0.0f;
        size_t row = (size_t)(b * 2048 + q0 + 8 * h + j) * 1024 + hh * 64;
        #pragma unroll
        for (int t = 0; t < 4; ++t)
            Ob[row + 16 * t + n] = f2bf(o[t][j] * inv);
    }
}

// ---------------------------------------------------------------------------
// Launcher
// ---------------------------------------------------------------------------
extern "C" void kernel_launch(void* const* d_in, const int* in_sizes, int n_in,
                              void* d_out, int out_size, void* d_ws, size_t ws_size,
                              hipStream_t stream) {
    const float* H    = (const float*)d_in[0];
    const int*   mask = (const int*)  d_in[1];
    const float* Wq   = (const float*)d_in[2];
    const float* Wk   = (const float*)d_in[3];
    const float* Wv   = (const float*)d_in[4];
    const float* Wo   = (const float*)d_in[5];
    float* out = (float*)d_out;

    const int M = 4096, N = 1024, K = 1024;
    const int HK = M * N;
    const int WK = N * K;

    unsigned short* ws  = (unsigned short*)d_ws;
    unsigned short* Hb  = ws;
    unsigned short* Wqb = Hb  + HK;
    unsigned short* Wkb = Wqb + WK;
    unsigned short* Wvb = Wkb + WK;
    unsigned short* Wob = Wvb + WK;
    unsigned short* Qb  = Wob + WK;
    unsigned short* Kbf = Qb  + HK;
    unsigned short* Vt  = Kbf + HK;   // [B*HEADS*64, 2048] transposed V
    unsigned short* Ob  = Vt  + HK;

    attn_f32_to_bf16<<<(HK + 255) / 256, 256, 0, stream>>>(H,  Hb,  HK);
    attn_f32_to_bf16<<<(WK + 255) / 256, 256, 0, stream>>>(Wq, Wqb, WK);
    attn_f32_to_bf16<<<(WK + 255) / 256, 256, 0, stream>>>(Wk, Wkb, WK);
    attn_f32_to_bf16<<<(WK + 255) / 256, 256, 0, stream>>>(Wv, Wvb, WK);
    attn_f32_to_bf16<<<(WK + 255) / 256, 256, 0, stream>>>(Wo, Wob, WK);

    dim3 gg(M / 64, N / 128);   // 64 x 8
    attn_gemm_bf16<<<gg, 256, 0, stream>>>(Hb, Wqb, Qb,  M, N, K, 0.125f, 1);
    attn_gemm_bf16<<<gg, 256, 0, stream>>>(Hb, Wkb, Kbf, M, N, K, 1.0f,   1);
    attn_gemm_bf16<<<gg, 256, 0, stream>>>(Hb, Wvb, Vt,  M, N, K, 1.0f,   2);

    attn_flash<<<dim3(32, 128), 32, 0, stream>>>(Qb, Kbf, Vt, mask, Ob);

    attn_gemm_bf16<<<gg, 256, 0, stream>>>(Ob, Wob, out, M, N, K, 1.0f, 0);
}
